// Model_979252543658
// MI455X (gfx1250) — matrix operand order repaired and checked by reading, not compiled, not run on hardware
//
#include <hip/hip_runtime.h>
#include <hip/hip_bf16.h>
#include <math.h>

// ---------------------------------------------------------------------------
// CDNA5 WMMA helpers (wave32, V_WMMA_F32_16X16X32_F16)
// ---------------------------------------------------------------------------
typedef __attribute__((ext_vector_type(16))) _Float16 v16h;
typedef __attribute__((ext_vector_type(8)))  float    v8f;

#define EPS 1e-5f

__device__ __forceinline__ float gelu_f(float x) {
    return 0.5f * x * (1.0f + erff(x * 0.70710678118654752f));
}

// A fragment: 16x32 f16, row-major LDS source A[m][k], stride ld (halves).
// lanes 0-15 hold K {k0..k0+15}; lanes 16-31 hold K {k0+16..k0+31}.
__device__ __forceinline__ v16h load_A(const _Float16* A, int m0, int k0, int ld) {
    int lane = threadIdx.x & 31;
    int m  = m0 + (lane & 15);
    int kh = k0 + ((lane >> 4) << 4);
    v16h a;
#pragma unroll
    for (int i = 0; i < 8; ++i) a[i]     = A[m * ld + kh + i];
#pragma unroll
    for (int i = 0; i < 8; ++i) a[8 + i] = A[m * ld + kh + 8 + i];
    return a;
}

// B fragment from N-major LDS buffer Bt[n][k] (stride ldk halves).
// lanes 0-15 hold K {k0..k0+15}; lanes 16-31 hold K {k0+16..k0+31}.
// Contiguous per-lane reads -> ds_load_b128 pairs.
__device__ __forceinline__ v16h load_Bt(const _Float16* Bt, int k0, int n0, int ldk) {
    int lane = threadIdx.x & 31;
    int n  = n0 + (lane & 15);
    int kb = k0 + ((lane >> 4) << 4);
    v16h b;
#pragma unroll
    for (int i = 0; i < 16; ++i) b[i] = Bt[n * ldk + kb + i];
    return b;
}

__device__ __forceinline__ v8f wmma32(v16h a, v16h b, v8f c) {
    return __builtin_amdgcn_wmma_f32_16x16x32_f16(false, a, false, b,
                                                  (short)0, c, false, false);
}

// C/D 16x16 f32: VGPR r holds M = r (lanes 0-15) or 8+r (lanes 16-31), N = lane&15.
__device__ __forceinline__ void store_C(float* C, v8f c, int m0, int n0, int ld) {
    int lane = threadIdx.x & 31;
    int n  = n0 + (lane & 15);
    int mb = m0 + ((lane >> 4) << 3);
#pragma unroll
    for (int r = 0; r < 8; ++r) C[(mb + r) * ld + n] = c[r];
}

__device__ __forceinline__ v8f load_C_global(const float* C, int m0, int n0, int ld) {
    int lane = threadIdx.x & 31;
    int n  = n0 + (lane & 15);
    int mb = m0 + ((lane >> 4) << 3);
    v8f c;
#pragma unroll
    for (int r = 0; r < 8; ++r) c[r] = C[(mb + r) * ld + n];
    return c;
}

// ---------------------------------------------------------------------------
// Fused feature kernel: conv1 -> pool(28->8) -> gelu -> conv2 -> conv3
// -> conv4 -> flatten -> linear. One block per image, 256 threads (8 waves).
// conv2/conv3 are WMMA GEMMs: M=64 pixels, N=32 channels, K=288 (im2col).
// ---------------------------------------------------------------------------
#define A_STRIDE 296   // 288 K + pad (halves); row = 592 B (16B multiple)
#define B_STRIDE 296   // N-major weights: 32 rows x 296 halves
#define SMEM_A_BYTES   (64 * A_STRIDE * 2)   // 37888
#define SMEM_B_BYTES   (32 * B_STRIDE * 2)   // 18944
#define SMEM_F_OFF     (SMEM_A_BYTES + SMEM_B_BYTES)
#define SMEM_TOTAL     (SMEM_F_OFF + 32 * 64 * 2)  // 60928 < 64KB

__device__ void conv_stage_wmma(_Float16* A, _Float16* Bw, _Float16* feat,
                                const float* w, const float* bias, const float* bn) {
    int tid = threadIdx.x;
    // Weights, N-major: Bw[n][k = ic*9 + r9] <- w[n][ic][r9] (9 contiguous halves)
    for (int pair = tid; pair < 32 * 32; pair += 256) {
        int n = pair >> 5, ic = pair & 31;
        const float* wp = w + (n * 32 + ic) * 9;
        _Float16* dst = Bw + n * B_STRIDE + ic * 9;
#pragma unroll
        for (int r9 = 0; r9 < 9; ++r9) dst[r9] = (_Float16)wp[r9];
    }
    // im2col activations A[p][ic*9 + r9], zero-padded border (9 contiguous halves)
    for (int pair = tid; pair < 64 * 32; pair += 256) {
        int p = pair >> 5, ic = pair & 31;
        int oy = p >> 3, ox = p & 7;
        _Float16* dst = A + p * A_STRIDE + ic * 9;
        const _Float16* fin = feat + ic * 64;
#pragma unroll
        for (int ky = 0; ky < 3; ++ky) {
            int y = oy + ky - 1;
#pragma unroll
            for (int kx = 0; kx < 3; ++kx) {
                int x = ox + kx - 1;
                _Float16 v = (_Float16)0.f;
                if ((unsigned)y < 8u && (unsigned)x < 8u) v = fin[y * 8 + x];
                dst[ky * 3 + kx] = v;
            }
        }
    }
    __syncthreads();

    int wv = tid >> 5;
    int mt = wv >> 1, nt = wv & 1;           // 4 x 2 tiles over 8 waves
    v8f acc = {};
#pragma unroll
    for (int kk = 0; kk < 9; ++kk) {
        v16h a = load_A(A, mt * 16, kk * 32, A_STRIDE);
        v16h b = load_Bt(Bw, kk * 32, nt * 16, B_STRIDE);
        acc = wmma32(a, b, acc);
    }
    __syncthreads();                          // acc in regs; safe to overwrite feat

    int lane = tid & 31;
    int n  = nt * 16 + (lane & 15);           // out channel
    int mb = mt * 16 + ((lane >> 4) << 3);    // pixel base
    float sc = bn[n] * rsqrtf(bn[96 + n] + EPS);
    float mu = bn[64 + n], bt = bn[32 + n], bsn = bias[n];
#pragma unroll
    for (int r = 0; r < 8; ++r) {
        float v = acc[r] + bsn;
        v = (v - mu) * sc + bt;
        feat[n * 64 + (mb + r)] = (_Float16)gelu_f(v);
    }
    __syncthreads();
}

__global__ __launch_bounds__(256) void feat_kernel(
    const float* __restrict__ sxs, const float* __restrict__ qxs,
    const float* __restrict__ c1w, const float* __restrict__ c1b, const float* __restrict__ bn1,
    const float* __restrict__ c2w, const float* __restrict__ c2b, const float* __restrict__ bn2,
    const float* __restrict__ c3w, const float* __restrict__ c3b, const float* __restrict__ bn3,
    const float* __restrict__ c4w, const float* __restrict__ c4b, const float* __restrict__ bn4,
    const float* __restrict__ lin_w, float* __restrict__ feats_out)
{
    __shared__ __align__(16) char smem[SMEM_TOTAL];
    _Float16* A    = (_Float16*)smem;
    _Float16* Bw   = (_Float16*)(smem + SMEM_A_BYTES);
    _Float16* feat = (_Float16*)(smem + SMEM_F_OFF);

    int img = blockIdx.x;
    int tid = threadIdx.x;
    const float* src = (img < 3072) ? (sxs + (size_t)img * 784)
                                    : (qxs + (size_t)(img - 3072) * 784);
    // ---- conv1 + pool + bn + gelu (alias A region for image + weights) ----
    float* imgS = (float*)smem;               // 784 floats
    float* wS   = (float*)(smem + 3200);      // 288 floats
    for (int i = tid; i < 784; i += 256) imgS[i] = src[i];
    for (int i = tid; i < 288; i += 256) wS[i] = c1w[i];
    __syncthreads();
#pragma unroll
    for (int j = 0; j < 8; ++j) {
        int out = tid + j * 256;              // 32ch x 64px
        int ch = out >> 6, p = out & 63;
        int oy = p >> 3, ox = p & 7;
        int sy = (oy * 28) >> 3, ey = ((oy + 1) * 28 + 7) >> 3;
        int sx = (ox * 28) >> 3, ex = ((ox + 1) * 28 + 7) >> 3;
        float acc = 0.f;
        for (int y = sy; y < ey; ++y)
            for (int x = sx; x < ex; ++x) {
                float cv = 0.f;
#pragma unroll
                for (int ky = 0; ky < 3; ++ky) {
                    int yy = y + ky - 1;
                    if ((unsigned)yy >= 28u) continue;
#pragma unroll
                    for (int kx = 0; kx < 3; ++kx) {
                        int xx = x + kx - 1;
                        if ((unsigned)xx >= 28u) continue;
                        cv += imgS[yy * 28 + xx] * wS[ch * 9 + ky * 3 + kx];
                    }
                }
                acc += cv;
            }
        float v = acc / (float)((ey - sy) * (ex - sx)) + c1b[ch];
        float sc = bn1[ch] * rsqrtf(bn1[96 + ch] + EPS);
        v = (v - bn1[64 + ch]) * sc + bn1[32 + ch];
        feat[ch * 64 + p] = (_Float16)gelu_f(v);
    }
    __syncthreads();

    // ---- conv2 / conv3 as WMMA GEMMs ----
    conv_stage_wmma(A, Bw, feat, c2w, c2b, bn2);
    conv_stage_wmma(A, Bw, feat, c3w, c3b, bn3);

    // ---- conv4 (32->1) + bn + gelu + linear (64x64) ----
    float* h4s = (float*)smem;                // reuse A region
    if (tid < 64) {
        int oy = tid >> 3, ox = tid & 7;
        float acc = c4b[0];
        for (int ic = 0; ic < 32; ++ic)
#pragma unroll
            for (int ky = 0; ky < 3; ++ky) {
                int yy = oy + ky - 1;
                if ((unsigned)yy >= 8u) continue;
#pragma unroll
                for (int kx = 0; kx < 3; ++kx) {
                    int xx = ox + kx - 1;
                    if ((unsigned)xx >= 8u) continue;
                    acc += (float)feat[ic * 64 + yy * 8 + xx] * c4w[ic * 9 + ky * 3 + kx];
                }
            }
        float sc = bn4[0] * rsqrtf(bn4[3] + EPS);
        acc = (acc - bn4[2]) * sc + bn4[1];
        h4s[tid] = gelu_f(acc);
    }
    __syncthreads();
    if (tid < 64) {
        float s = 0.f;
        for (int d = 0; d < 64; ++d) s += h4s[d] * lin_w[tid * 64 + d];
        feats_out[(size_t)img * 64 + tid] = s;
    }
}

// ---------------------------------------------------------------------------
// Fused per-task kernel: inp build -> _net (2x WMMA GEMM) -> attention ->
// r_o projection (WMMA) -> layernorm -> rank-6 weight updates (WMMA, K pad 32)
// -> query net -> logits -> CE loss / accuracy atomics.
// One block per task (512 blocks), 256 threads (8 waves).
// sB is N-major: sB[n][k], stride LD (row = 144 B, 16B multiple).
// ---------------------------------------------------------------------------
#define LD 72

__global__ __launch_bounds__(256) void lor_kernel(
    const float* __restrict__ feats, const float* __restrict__ emb,
    const float* __restrict__ w1, const float* __restrict__ w2,
    const float* __restrict__ tags,
    const float* __restrict__ r_q, const float* __restrict__ r_k,
    const float* __restrict__ r_v, const float* __restrict__ r_o,
    const float* __restrict__ ln_w, const float* __restrict__ ln_b,
    const int* __restrict__ sys, const int* __restrict__ qys,
    float* __restrict__ lossAcc, unsigned int* __restrict__ corrAcc)
{
    __shared__ _Float16 sA[64 * LD];     // A fragments (f16), row-major [m][k]
    __shared__ _Float16 sB[64 * LD];     // B fragments (f16), N-major [n][k]
    __shared__ float    sX[32 * 64];     // inp (residual)
    __shared__ float    sH[32 * 64];     // GEMM results
    __shared__ float    sL[32 * 64];     // lor / lors
    __shared__ float    sQW[64 * 64];    // qw1 / qw2
    __shared__ float    sPA[2 * 64];     // gelu(hq)
    __shared__ float    sPred[2 * 64];   // preds

    int b = blockIdx.x;
    int tid = threadIdx.x;
    __builtin_prefetch(w1, 0, 0);
    __builtin_prefetch(w2, 0, 0);

    // ---- 1. inp[k][d] = sx_feat + emb[sys] + tags[t], rows 24..31 zero ----
    for (int idx = tid; idx < 32 * 64; idx += 256) {
        int k = idx >> 6, d = idx & 63;
        float v = 0.f;
        if (k < 24) {
            int s = k >> 2, t = k & 3;
            int lbl = sys[b * 6 + s];
            v = feats[(size_t)(b * 6 + s) * 64 + d] + emb[lbl * 64 + d] + tags[t * 64 + d];
        }
        sX[idx] = v;
        sA[k * LD + d] = (_Float16)v;
    }
    // B = w1 (row-major [e][d] is already N-major [n=e][k=d]: straight copy)
    for (int idx = tid; idx < 64 * 64; idx += 256) {
        int e = idx >> 6, d = idx & 63;
        sB[e * LD + d] = (_Float16)w1[idx];
    }
    __syncthreads();

    // ---- 2. h = inp @ w1^T  (M=32, N=64, K=64) ----
    {
        int wv = tid >> 5, mt = wv >> 2, nt = wv & 3;
        v8f acc = {};
#pragma unroll
        for (int kk = 0; kk < 2; ++kk)
            acc = wmma32(load_A(sA, mt * 16, kk * 32, LD),
                         load_Bt(sB, kk * 32, nt * 16, LD), acc);
        store_C(sH, acc, mt * 16, nt * 16, 64);
    }
    __syncthreads();

    // ---- 3. ah = gelu(h); B = w2 ----
    for (int idx = tid; idx < 32 * 64; idx += 256) {
        int k = idx >> 6, d = idx & 63;
        sA[k * LD + d] = (_Float16)((k < 24) ? gelu_f(sH[idx]) : 0.f);
    }
    for (int idx = tid; idx < 64 * 64; idx += 256) {
        int e = idx >> 6, d = idx & 63;
        sB[e * LD + d] = (_Float16)w2[idx];
    }
    __syncthreads();

    // ---- 4. q = ah @ w2^T ----
    {
        int wv = tid >> 5, mt = wv >> 2, nt = wv & 3;
        v8f acc = {};
#pragma unroll
        for (int kk = 0; kk < 2; ++kk)
            acc = wmma32(load_A(sA, mt * 16, kk * 32, LD),
                         load_Bt(sB, kk * 32, nt * 16, LD), acc);
        store_C(sH, acc, mt * 16, nt * 16, 64);
    }
    __syncthreads();

    // ---- 5. attention over 16 slots: thread = (k, head) ----
    if (tid < 96) {
        int k = tid >> 2, n = tid & 3;
        float qh[16];
#pragma unroll
        for (int h = 0; h < 16; ++h) {
            float s = 0.f;
            const float* rq = r_q + (n * 16 + h) * 64;
            for (int d = 0; d < 64; ++d) s += rq[d] * sH[k * 64 + d];
            qh[h] = s;
        }
        float scv[16], mx = -1e30f;
#pragma unroll
        for (int r = 0; r < 16; ++r) {
            float s = 0.f;
            const float* rk = r_k + (r * 4 + n) * 16;
#pragma unroll
            for (int h = 0; h < 16; ++h) s += qh[h] * rk[h];
            s *= 0.25f;                       // 1/sqrt(HD)
            scv[r] = s;
            mx = fmaxf(mx, s);
        }
        float den = 0.f;
#pragma unroll
        for (int r = 0; r < 16; ++r) { scv[r] = __expf(scv[r] - mx); den += scv[r]; }
        float inv = 1.f / den;
#pragma unroll
        for (int h = 0; h < 16; ++h) {
            float o = 0.f;
#pragma unroll
            for (int r = 0; r < 16; ++r) o += scv[r] * r_v[(r * 4 + n) * 16 + h];
            sL[k * 64 + n * 16 + h] = o * inv;
        }
    }
    __syncthreads();

    // ---- 6. lo = lor @ r_o  (B[n=e][k=d] = r_o[d][e]) ----
    for (int idx = tid; idx < 32 * 64; idx += 256) {
        int k = idx >> 6, d = idx & 63;
        sA[k * LD + d] = (_Float16)((k < 24) ? sL[idx] : 0.f);
    }
    for (int idx = tid; idx < 64 * 64; idx += 256) {
        int e = idx >> 6, d = idx & 63;
        sB[e * LD + d] = (_Float16)r_o[d * 64 + e];
    }
    __syncthreads();
    {
        int wv = tid >> 5, mt = wv >> 2, nt = wv & 3;
        v8f acc = {};
#pragma unroll
        for (int kk = 0; kk < 2; ++kk)
            acc = wmma32(load_A(sA, mt * 16, kk * 32, LD),
                         load_Bt(sB, kk * 32, nt * 16, LD), acc);
        store_C(sH, acc, mt * 16, nt * 16, 64);
    }
    __syncthreads();

    // ---- 7. z = lo + inp; layernorm -> lors (into sL) ----
    if (tid < 24) {
        float mu = 0.f;
        for (int d = 0; d < 64; ++d) mu += sH[tid * 64 + d] + sX[tid * 64 + d];
        mu *= (1.f / 64.f);
        float var = 0.f;
        for (int d = 0; d < 64; ++d) {
            float z = sH[tid * 64 + d] + sX[tid * 64 + d] - mu;
            var += z * z;
        }
        var *= (1.f / 64.f);
        float inv = rsqrtf(var + EPS);
        for (int d = 0; d < 64; ++d) {
            float z = sH[tid * 64 + d] + sX[tid * 64 + d] - mu;
            sL[tid * 64 + d] = z * inv * ln_w[d] + ln_b[d];
        }
    }
    __syncthreads();

    // ---- 8/9. two passes: qw1 + hidden, then qw2 + preds ----
    for (int pass = 0; pass < 2; ++pass) {
        const float* wbase = pass ? w2 : w1;
        // A[l][s] = l-vec (rank-6, K padded to 32); B[n=r][k=s] = r-vec
        for (int idx = tid; idx < 64 * 32; idx += 256) {
            int l = idx >> 5, ss = idx & 31;
            sA[l * LD + ss] = (ss < 6) ? (_Float16)sL[(ss * 4 + 2 * pass) * 64 + l]
                                       : (_Float16)0.f;
        }
        for (int idx = tid; idx < 64 * 32; idx += 256) {
            int r = idx >> 5, ss = idx & 31;
            sB[r * LD + ss] = (ss < 6) ? (_Float16)sL[(ss * 4 + 2 * pass + 1) * 64 + r]
                                       : (_Float16)0.f;
        }
        __syncthreads();
        {
            int wv = tid >> 5;
#pragma unroll
            for (int t = 0; t < 2; ++t) {      // 16 tiles over 8 waves
                int tile = wv * 2 + t;
                int mt = tile >> 2, nt = tile & 3;
                v8f acc = load_C_global(wbase, mt * 16, nt * 16, 64);
                acc = wmma32(load_A(sA, mt * 16, 0, LD),
                             load_Bt(sB, 0, nt * 16, LD), acc);
                store_C(sQW, acc, mt * 16, nt * 16, 64);
            }
        }
        __syncthreads();
        if (tid < 128) {
            int qq = tid >> 6, e = tid & 63;
            float s = 0.f;
            if (pass == 0) {
                const float* qx = feats + (size_t)(3072 + b * 2 + qq) * 64;
                const float* qe = tags + 4 * 64;
                for (int d = 0; d < 64; ++d) s += (qx[d] + qe[d]) * sQW[e * 64 + d];
                sPA[qq * 64 + e] = gelu_f(s);
            } else {
                for (int d = 0; d < 64; ++d) s += sPA[qq * 64 + d] * sQW[e * 64 + d];
                sPred[qq * 64 + e] = s;
            }
        }
        __syncthreads();
    }

    // ---- 10. logits, CE loss, accuracy ----
    if (tid < 2) {
        int qq = tid;
        float lg[2];
#pragma unroll
        for (int l = 0; l < 2; ++l) {
            float s = 0.f;
            for (int d = 0; d < 64; ++d) s += emb[l * 64 + d] * sPred[qq * 64 + d];
            lg[l] = s;
        }
        int lbl = qys[b * 2 + qq];
        float mx = fmaxf(lg[0], lg[1]);
        float lse = mx + logf(__expf(lg[0] - mx) + __expf(lg[1] - mx));
        atomicAdd(lossAcc, lse - lg[lbl]);
        int pred = (lg[1] > lg[0]) ? 1 : 0;
        if (pred == lbl) atomicAdd(corrAcc, 1u);
    }
}

// ---------------------------------------------------------------------------
__global__ void zero_kernel(float* lossAcc, unsigned int* corrAcc) {
    if (threadIdx.x == 0) { lossAcc[0] = 0.f; corrAcc[0] = 0u; }
}

__global__ void finalize_kernel(const float* lossAcc, const unsigned int* corrAcc,
                                float* out) {
    float loss = lossAcc[0] * (1.0f / 1024.0f);
    out[0] = loss;
    out[1] = loss;
    out[2] = 0.0f;
    out[3] = (float)corrAcc[0];
    out[4] = 1024.0f;
}

// ---------------------------------------------------------------------------
extern "C" void kernel_launch(void* const* d_in, const int* in_sizes, int n_in,
                              void* d_out, int out_size, void* d_ws, size_t ws_size,
                              hipStream_t stream) {
    (void)in_sizes; (void)n_in; (void)out_size; (void)ws_size;
    const float* sxs   = (const float*)d_in[1];
    const float* qxs   = (const float*)d_in[2];
    const float* c1w   = (const float*)d_in[4];
    const float* c1b   = (const float*)d_in[5];
    const float* bn1   = (const float*)d_in[6];
    const float* c2w   = (const float*)d_in[7];
    const float* c2b   = (const float*)d_in[8];
    const float* bn2   = (const float*)d_in[9];
    const float* c3w   = (const float*)d_in[10];
    const float* c3b   = (const float*)d_in[11];
    const float* bn3   = (const float*)d_in[12];
    const float* c4w   = (const float*)d_in[13];
    const float* c4b   = (const float*)d_in[14];
    const float* bn4   = (const float*)d_in[15];
    const float* lin_w = (const float*)d_in[16];
    const float* emb   = (const float*)d_in[17];
    const float* w1    = (const float*)d_in[18];
    const float* w2    = (const float*)d_in[19];
    const float* tags  = (const float*)d_in[20];
    const float* r_q   = (const float*)d_in[21];
    const float* r_k   = (const float*)d_in[22];
    const float* r_v   = (const float*)d_in[23];
    const float* r_o   = (const float*)d_in[24];
    const float* ln_w  = (const float*)d_in[25];
    const float* ln_b  = (const float*)d_in[26];
    const int*   sys   = (const int*)d_in[27];
    const int*   qys   = (const int*)d_in[28];

    float* feats = (float*)d_ws;                          // 4096*64 f32 = 1 MB
    float* lossAcc = (float*)((char*)d_ws + (size_t)4096 * 64 * sizeof(float));
    unsigned int* corrAcc = (unsigned int*)(lossAcc + 1);

    zero_kernel<<<1, 32, 0, stream>>>(lossAcc, corrAcc);

    feat_kernel<<<4096, 256, 0, stream>>>(sxs, qxs,
                                          c1w, c1b, bn1, c2w, c2b, bn2,
                                          c3w, c3b, bn3, c4w, c4b, bn4,
                                          lin_w, feats);

    lor_kernel<<<512, 256, 0, stream>>>(feats, emb, w1, w2, tags,
                                        r_q, r_k, r_v, r_o, ln_w, ln_b,
                                        sys, qys, lossAcc, corrAcc);

    finalize_kernel<<<1, 1, 0, stream>>>(lossAcc, corrAcc, (float*)d_out);
}
